// MultiHeadLatentAttention_60644938219685
// MI455X (gfx1250) — matrix operand order repaired and checked by reading, not compiled, not run on hardware
//
#include <hip/hip_runtime.h>
#include <stdint.h>
#include <stddef.h>

// ---------------- problem constants ----------------
#define Bb   2
#define Ss   1024
#define Dd   2048
#define Hh   16
#define DN   128
#define DR   64
#define DV   128
#define RQ   1536
#define RKV  512
#define BS   (Bb*Ss)          // 2048 token rows
#define QKD  (DN+DR)          // 192
#define KVD  (DN+DV)          // 256
#define SCALE_F 0.07216878365f // (192)^-0.5

// ---------------- CDNA5 async global->LDS staging (guarded) ----------------
#if defined(__has_builtin)
#  if __has_builtin(__builtin_amdgcn_global_load_async_to_lds_b128)
#    define HAVE_ASYNC_LDS 1
#  endif
#endif
#ifndef HAVE_ASYNC_LDS
#  define HAVE_ASYNC_LDS 0
#endif

typedef int v4i __attribute__((ext_vector_type(4)));

static __device__ __forceinline__ void wait_async0() {
#if defined(__has_builtin) && __has_builtin(__builtin_amdgcn_s_wait_asynccnt)
    __builtin_amdgcn_s_wait_asynccnt(0);
#else
    asm volatile("s_wait_asynccnt 0x0" ::: "memory");
#endif
}

#if HAVE_ASYNC_LDS
// async DMA of 16 bytes: global -> LDS, no VGPR round trip (ASYNCcnt tracked)
static __device__ __forceinline__ void async_cp16(const unsigned short* gsrc,
                                                  unsigned short* ldst) {
    __builtin_amdgcn_global_load_async_to_lds_b128(
        (v4i*)gsrc, (v4i*)ldst, 0, 0);
}
#endif

// ---------------- types ----------------
typedef float  v8f  __attribute__((ext_vector_type(8)));
typedef __bf16 bf16_t;
typedef bf16_t v16bf __attribute__((ext_vector_type(16)));
typedef bf16_t v8bf  __attribute__((ext_vector_type(8)));

static __device__ __forceinline__ unsigned short f2bf(float f) {
    unsigned int x = __float_as_uint(f);
    unsigned int r = (x + 0x7FFFu + ((x >> 16) & 1u)) >> 16; // RNE
    return (unsigned short)r;
}

static __device__ __forceinline__ v8f v8f_zero() {
    v8f v;
#pragma unroll
    for (int i = 0; i < 8; ++i) v[i] = 0.0f;
    return v;
}

static __device__ __forceinline__ v16bf mk16(v8bf lo, v8bf hi) {
    union { v16bf v; v8bf h[2]; } u;
    u.h[0] = lo; u.h[1] = hi;
    return u.v;
}
static __device__ __forceinline__ v8bf ld8(const unsigned short* p) {
    return *reinterpret_cast<const v8bf*>(p);
}

// ---------------- fp32 -> bf16 convert (elementwise, for x) ----------------
__global__ void k_f32_to_bf16(const float* __restrict__ src,
                              unsigned short* __restrict__ dst, long n) {
    long i = (long)blockIdx.x * blockDim.x + threadIdx.x;
    long stride = (long)gridDim.x * blockDim.x;
    for (; i < n; i += stride) dst[i] = f2bf(src[i]);
}

// ---------------- fp32 -> bf16 convert + transpose (for weights) ----------------
// W [K][N] fp32 (row-major)  ->  WT [N][K] bf16 (row-major)
__global__ __launch_bounds__(256)
void k_transpose_bf16(const float* __restrict__ W, unsigned short* __restrict__ WT,
                      int K, int N) {
    __shared__ float tile[32][33];
    const int kb = blockIdx.y * 32, nb = blockIdx.x * 32;
    const int r  = threadIdx.x >> 3;        // 0..31
    const int c4 = (threadIdx.x & 7) * 4;   // 0,4,..,28
#pragma unroll
    for (int i = 0; i < 4; ++i) {
        int k = kb + r, n = nb + c4 + i;
        tile[r][c4 + i] = (k < K && n < N) ? W[(size_t)k * N + n] : 0.0f;
    }
    __syncthreads();
#pragma unroll
    for (int i = 0; i < 4; ++i) {
        int n = nb + r, k = kb + c4 + i;
        if (n < N && k < K) WT[(size_t)n * K + k] = f2bf(tile[c4 + i][r]);
    }
}

// ---------------- RMSNorm (fp32 in, bf16 out) ----------------
__global__ __launch_bounds__(256)
void k_rmsnorm_bf16(const float* __restrict__ X, const float* __restrict__ W,
                    unsigned short* __restrict__ Y, int cols, int ldx) {
    int row = blockIdx.x;
    const float* x = X + (size_t)row * ldx;
    __shared__ float red[256];
    float ss = 0.f;
    for (int c = threadIdx.x; c < cols; c += 256) { float v = x[c]; ss += v * v; }
    red[threadIdx.x] = ss;
    __syncthreads();
    for (int s = 128; s > 0; s >>= 1) {
        if ((int)threadIdx.x < s) red[threadIdx.x] += red[threadIdx.x + s];
        __syncthreads();
    }
    float scale = rsqrtf(red[0] / (float)cols + 1e-6f);
    for (int c = threadIdx.x; c < cols; c += 256)
        Y[(size_t)row * cols + c] = f2bf(x[c] * scale * W[c]);
}

// ---------------- bf16 WMMA GEMM: C[M,N] = A[M,K] @ BT[N,K]^T ----------------
// A row-major bf16 [M,K]; BT row-major bf16 [N,K] (pre-transposed weight).
// C fp32 [M,N]. Requires: M % 128 == 0, K % 32 == 0, N % 16 == 0.
// Block 256 threads = 8 waves (4x2), tile 128x128, K-step 32.
__global__ __launch_bounds__(256)
void k_gemm_bf16(const unsigned short* __restrict__ A,
                 const unsigned short* __restrict__ BT,
                 float* __restrict__ C, int M, int N, int K) {
    __shared__ unsigned short As [128][40];  // padded: 80B row stride, 16B aligned
    __shared__ unsigned short BTs[128][40];  // [n][k]

    const int tid  = threadIdx.x;
    const int lane = tid & 31;
    const int wave = tid >> 5;
    const int wm   = wave & 3;           // wave row  (4)
    const int wn   = wave >> 2;          // wave col  (2)
    const int hi   = (lane >> 4) & 1;
    const int l16  = lane & 15;
    const int mbase = blockIdx.y * 128;
    const int nbase = blockIdx.x * 128;

    v8f acc[2][4];
#pragma unroll
    for (int im = 0; im < 2; ++im)
#pragma unroll
        for (int in = 0; in < 4; ++in) acc[im][in] = v8f_zero();

    const int lrow = tid >> 1, lcb = (tid & 1) * 16;  // loader: 16 elems/thread
    const bool bvalid = (nbase + lrow) < N;           // uniform per thread, whole row

    const unsigned short* aptr = A  + (size_t)(mbase + lrow) * K + lcb;
    const unsigned short* bptr = BT + (size_t)(nbase + lrow) * K + lcb;

    const int ksteps = K >> 5;
    for (int kt = 0; kt < ksteps; ++kt) {
        // ---- stage A tile 128x32 and B^T tile 128x32 ----
#if HAVE_ASYNC_LDS
        // CDNA5 async DMA: global -> LDS without a VGPR round trip (ASYNCcnt).
        async_cp16(aptr,     &As[lrow][lcb]);
        async_cp16(aptr + 8, &As[lrow][lcb + 8]);
        if (bvalid) {
            async_cp16(bptr,     &BTs[lrow][lcb]);
            async_cp16(bptr + 8, &BTs[lrow][lcb + 8]);
        }
        wait_async0();
#else
        *reinterpret_cast<v8bf*>(&As[lrow][lcb])     = ld8(aptr);
        *reinterpret_cast<v8bf*>(&As[lrow][lcb + 8]) = ld8(aptr + 8);
        if (bvalid) {
            *reinterpret_cast<v8bf*>(&BTs[lrow][lcb])     = ld8(bptr);
            *reinterpret_cast<v8bf*>(&BTs[lrow][lcb + 8]) = ld8(bptr + 8);
        }
#endif
        if (kt + 1 < ksteps) {
            __builtin_prefetch(aptr + 32, 0, 1);
            __builtin_prefetch(bptr + 32, 0, 1);
        }
        aptr += 32;
        bptr += 32;
        __syncthreads();

        // ---- fragments ----
        v16bf afrag[2], bfrag[4];
#pragma unroll
        for (int im = 0; im < 2; ++im) {
            const unsigned short* p = &As[wm * 32 + im * 16 + l16][hi * 8];
            afrag[im] = mk16(ld8(p), ld8(p + 16)); // K runs {0..7,16..23}/{8..15,24..31}
        }
#pragma unroll
        for (int in = 0; in < 4; ++in) {
            const unsigned short* p = &BTs[wn * 64 + in * 16 + l16][hi * 8];
            bfrag[in] = mk16(ld8(p), ld8(p + 16)); // K runs {0..7,16..23}/{8..15,24..31}
        }
#pragma unroll
        for (int im = 0; im < 2; ++im)
#pragma unroll
            for (int in = 0; in < 4; ++in)
                acc[im][in] = __builtin_amdgcn_wmma_f32_16x16x32_bf16(
                    false, afrag[im], false, bfrag[in], (short)0, acc[im][in],
                    false, false);
        __syncthreads();
    }

    // ---- epilogue: C fp32 (N % 16 == 0 -> whole 16-col group valid or not) ----
#pragma unroll
    for (int in = 0; in < 4; ++in) {
        const int colg = nbase + wn * 64 + in * 16;
        if (colg >= N) continue;           // uniform scalar branch
#pragma unroll
        for (int im = 0; im < 2; ++im)
#pragma unroll
            for (int r = 0; r < 8; ++r) {
                int row = mbase + wm * 32 + im * 16 + r + hi * 8;
                C[(size_t)row * N + colg + l16] = acc[im][in][r];
            }
    }
}

// ---------------- RoPE helpers ----------------
// inv_freq[j] = 10000^(-2j/64) = exp(-j * ln(10000)/32); ln(10000)/32 = 0.28782313662f
static __device__ __forceinline__ void rope_cs(int i, int s, float& c, float& sn) {
    int j = i & 31;
    float inv = __expf(-(float)j * 0.28782313662f);
    float ang = (float)s * inv;
    c  = __cosf(ang);
    sn = __sinf(ang);
}

// ---------------- assemble q_final [b,h,s,192] bf16 ----------------
__global__ void k_assemble_q(const float* __restrict__ Q,
                             unsigned short* __restrict__ qf) {
    int idx = blockIdx.x;            // (b*S+s)*H + h
    int h = idx % Hh; int bs = idx / Hh;
    int s = bs % Ss;  int b = bs / Ss;
    int d = threadIdx.x;             // 0..191
    const float* src = Q + (size_t)bs * (Hh * QKD) + h * QKD;
    float v = src[d];
    if (d >= DN) {
        int i = d - DN;
        float c, sn; rope_cs(i, s, c, sn);
        float partner = (i & 1) ? src[d - 1] : src[d + 1];
        float rot = (i & 1) ? partner : -partner;
        v = v * c + rot * sn;
    }
    qf[((size_t)(b * Hh + h) * Ss + s) * QKD + d] = f2bf(v);
}

// ---------------- assemble k_final [b,h,s,192] and V^T [b,h,dv,s] ----------------
__global__ void k_assemble_kv(const float* __restrict__ KV,
                              const float* __restrict__ KVA,
                              unsigned short* __restrict__ kf,
                              unsigned short* __restrict__ vt) {
    int idx = blockIdx.x;            // (b*S+s)*H + h
    int h = idx % Hh; int bs = idx / Hh;
    int s = bs % Ss;  int b = bs / Ss;
    int t = threadIdx.x;             // 0..255
    const float* kvrow = KV + (size_t)bs * (Hh * KVD) + h * KVD;
    size_t kbase = ((size_t)(b * Hh + h) * Ss + s) * QKD;
    if (t < DN) {
        kf[kbase + t] = f2bf(kvrow[t]);                                     // k_nope
        vt[((size_t)(b * Hh + h) * DV + t) * Ss + s] = f2bf(kvrow[DN + t]); // V^T
    } else if (t < DN + DR) {
        int i = t - DN;
        const float* kr = KVA + (size_t)bs * (RKV + DR) + RKV;              // shared rope
        float x = kr[i];
        float c, sn; rope_cs(i, s, c, sn);
        float partner = (i & 1) ? kr[i - 1] : kr[i + 1];
        float rot = (i & 1) ? partner : -partner;
        kf[kbase + DN + i] = f2bf(x * c + rot * sn);
    }
}

// ---------------- causal flash attention, one 16-query tile per wave ----------------
// grid: B*H*8 blocks of 256 threads (8 waves -> 128 query rows / block)
__global__ __launch_bounds__(256)
void k_mla_attention(const unsigned short* __restrict__ qf,
                     const unsigned short* __restrict__ kf,
                     const unsigned short* __restrict__ vt,
                     unsigned short* __restrict__ attn) {
    __shared__ unsigned short Pbuf[8][16][32]; // per-wave P staging (layout swap)

    const int wave = threadIdx.x >> 5;
    const int lane = threadIdx.x & 31;
    const int hi   = lane >> 4;
    const int l16  = lane & 15;

    const int tile8 = blockIdx.x & 7;
    const int bh    = blockIdx.x >> 3;        // 0..31
    const int b = bh >> 4, h = bh & 15;
    const int qbase = (tile8 * 8 + wave) * 16;

    const unsigned short* Q  = qf + (size_t)bh * Ss * QKD;
    const unsigned short* Kf = kf + (size_t)bh * Ss * QKD;
    const unsigned short* V  = vt + (size_t)bh * DV * Ss;

    v8f o[8];
#pragma unroll
    for (int t = 0; t < 8; ++t) o[t] = v8f_zero();
    float m[8], l[8];
#pragma unroll
    for (int r = 0; r < 8; ++r) { m[r] = -3.0e38f; l[r] = 0.0f; }

    for (int kb = 0; kb < qbase + 16; kb += 32) {
        // ---- scores: Q[16x192] @ K^T[192x32] via 12 WMMAs ----
        v8f s0 = v8f_zero(), s1 = v8f_zero();
#pragma unroll
        for (int kc = 0; kc < 6; ++kc) {
            const unsigned short* qp =
                Q + (size_t)(qbase + l16) * QKD + kc * 32 + hi * 8;
            v16bf aq = mk16(ld8(qp), ld8(qp + 16));
            const unsigned short* kp0 =
                Kf + (size_t)(kb + l16) * QKD + kc * 32 + hi * 8;
            v16bf bk0 = mk16(ld8(kp0), ld8(kp0 + 16));
            s0 = __builtin_amdgcn_wmma_f32_16x16x32_bf16(
                false, aq, false, bk0, (short)0, s0, false, false);
            const unsigned short* kp1 =
                Kf + (size_t)(kb + 16 + l16) * QKD + kc * 32 + hi * 8;
            v16bf bk1 = mk16(ld8(kp1), ld8(kp1 + 16));
            s1 = __builtin_amdgcn_wmma_f32_16x16x32_bf16(
                false, aq, false, bk1, (short)0, s1, false, false);
        }

        // ---- scale + causal mask + online softmax ----
        const int key0 = kb + l16, key1 = kb + 16 + l16;
        float alpha[8];
#pragma unroll
        for (int r = 0; r < 8; ++r) {
            int qrow = qbase + r + hi * 8;
            float v0 = (key0 > qrow) ? -1.0e9f : s0[r] * SCALE_F;
            float v1 = (key1 > qrow) ? -1.0e9f : s1[r] * SCALE_F;
            float mx = fmaxf(v0, v1);
#pragma unroll
            for (int msk = 8; msk >= 1; msk >>= 1)
                mx = fmaxf(mx, __shfl_xor(mx, msk, 32));
            float nm = fmaxf(m[r], mx);
            alpha[r] = __expf(m[r] - nm);
            m[r] = nm;
            float p0 = __expf(v0 - nm), p1 = __expf(v1 - nm);
            float sum = p0 + p1;
#pragma unroll
            for (int msk = 8; msk >= 1; msk >>= 1)
                sum += __shfl_xor(sum, msk, 32);
            l[r] = l[r] * alpha[r] + sum;
            Pbuf[wave][r + hi * 8][l16]      = f2bf(p0);
            Pbuf[wave][r + hi * 8][16 + l16] = f2bf(p1);
        }
#pragma unroll
        for (int t = 0; t < 8; ++t)
#pragma unroll
            for (int r = 0; r < 8; ++r) o[t][r] = o[t][r] * alpha[r];

        // ---- PV: P[16x32] @ V[32x128] via 8 WMMAs (same-wave DS ordering) ----
        const unsigned short* pp = &Pbuf[wave][l16][hi * 8];
        v16bf pa = mk16(ld8(pp), ld8(pp + 16));
#pragma unroll
        for (int t = 0; t < 8; ++t) {
            const unsigned short* vp =
                V + (size_t)(t * 16 + l16) * Ss + kb + hi * 8;
            v16bf bv = mk16(ld8(vp), ld8(vp + 16));
            o[t] = __builtin_amdgcn_wmma_f32_16x16x32_bf16(
                false, pa, false, bv, (short)0, o[t], false, false);
        }
    }

    // ---- epilogue: attn_bf[b, s, h*128 + dv] ----
#pragma unroll
    for (int r = 0; r < 8; ++r) {
        float inv = 1.0f / l[r];
        int srow = qbase + r + hi * 8;
        size_t rowoff = ((size_t)(b * Ss + srow)) * (Hh * DV) + h * DV;
#pragma unroll
        for (int t = 0; t < 8; ++t)
            attn[rowoff + t * 16 + l16] = f2bf(o[t][r] * inv);
    }
}

// ---------------- host-side launcher ----------------
extern "C" void kernel_launch(void* const* d_in, const int* in_sizes, int n_in,
                              void* d_out, int out_size, void* d_ws, size_t ws_size,
                              hipStream_t stream) {
    (void)in_sizes; (void)n_in; (void)out_size; (void)ws_size;
    const float* x        = (const float*)d_in[0];
    // d_in[1] = attention_mask (exact causal -1e9; implemented analytically)
    const float* Wqa      = (const float*)d_in[2];
    const float* qa_nw    = (const float*)d_in[3];
    const float* Wqb      = (const float*)d_in[4];
    const float* Wkva     = (const float*)d_in[5];
    const float* kva_nw   = (const float*)d_in[6];
    const float* Wkvb     = (const float*)d_in[7];
    const float* Wout     = (const float*)d_in[8];
    float* out = (float*)d_out;

    char* p = (char*)d_ws;
    auto alloc = [&](size_t bytes) -> void* {
        void* r = (void*)p;
        p += (bytes + 255) & ~(size_t)255;
        return r;
    };
    unsigned short* xb      = (unsigned short*)alloc((size_t)BS * Dd * 2);
    unsigned short* wqa_t   = (unsigned short*)alloc((size_t)Dd * RQ * 2);         // [RQ][D]
    unsigned short* wqb_t   = (unsigned short*)alloc((size_t)RQ * Hh * QKD * 2);   // [H*192][RQ]
    unsigned short* wkva_t  = (unsigned short*)alloc((size_t)Dd * (RKV + DR) * 2); // [576][D]
    unsigned short* wkvb_t  = (unsigned short*)alloc((size_t)RKV * Hh * KVD * 2);  // [H*256][RKV]
    unsigned short* wout_t  = (unsigned short*)alloc((size_t)Hh * DV * Dd * 2);    // [D][H*DV]
    float*          qa_f    = (float*)alloc((size_t)BS * RQ * 4);
    unsigned short* qa_bf   = (unsigned short*)alloc((size_t)BS * RQ * 2);
    float*          q_f     = (float*)alloc((size_t)BS * Hh * QKD * 4);
    float*          kva_f   = (float*)alloc((size_t)BS * (RKV + DR) * 4);
    unsigned short* kv_bf   = (unsigned short*)alloc((size_t)BS * RKV * 2);
    float*          kv_f    = (float*)alloc((size_t)BS * Hh * KVD * 4);
    unsigned short* qfin    = (unsigned short*)alloc((size_t)Bb * Hh * Ss * QKD * 2);
    unsigned short* kfin    = (unsigned short*)alloc((size_t)Bb * Hh * Ss * QKD * 2);
    unsigned short* vt      = (unsigned short*)alloc((size_t)Bb * Hh * DV * Ss * 2);
    unsigned short* attnb   = (unsigned short*)alloc((size_t)BS * Hh * DV * 2);

    // 1) bf16 conversions: x elementwise; weights converted + transposed to [N][K]
    k_f32_to_bf16<<<2048, 256, 0, stream>>>(x, xb, (long)BS * Dd);
    k_transpose_bf16<<<dim3(RQ / 32, Dd / 32), 256, 0, stream>>>(Wqa, wqa_t, Dd, RQ);
    k_transpose_bf16<<<dim3((Hh * QKD) / 32, RQ / 32), 256, 0, stream>>>(Wqb, wqb_t, RQ, Hh * QKD);
    k_transpose_bf16<<<dim3((RKV + DR) / 32, Dd / 32), 256, 0, stream>>>(Wkva, wkva_t, Dd, RKV + DR);
    k_transpose_bf16<<<dim3((Hh * KVD) / 32, RKV / 32), 256, 0, stream>>>(Wkvb, wkvb_t, RKV, Hh * KVD);
    k_transpose_bf16<<<dim3(Dd / 32, (Hh * DV) / 32), 256, 0, stream>>>(Wout, wout_t, Hh * DV, Dd);

    // 2) q_a = x @ Wqa ; rmsnorm -> bf16
    k_gemm_bf16<<<dim3((RQ + 127) / 128, BS / 128), 256, 0, stream>>>(
        xb, wqa_t, qa_f, BS, RQ, Dd);
    k_rmsnorm_bf16<<<BS, 256, 0, stream>>>(qa_f, qa_nw, qa_bf, RQ, RQ);

    // 3) q = qa_norm @ Wqb
    k_gemm_bf16<<<dim3((Hh * QKD + 127) / 128, BS / 128), 256, 0, stream>>>(
        qa_bf, wqb_t, q_f, BS, Hh * QKD, RQ);

    // 4) kv_a = x @ Wkva ; rmsnorm first 512 -> bf16
    k_gemm_bf16<<<dim3((RKV + DR + 127) / 128, BS / 128), 256, 0, stream>>>(
        xb, wkva_t, kva_f, BS, RKV + DR, Dd);
    k_rmsnorm_bf16<<<BS, 256, 0, stream>>>(kva_f, kva_nw, kv_bf, RKV, RKV + DR);

    // 5) kv = kv_norm @ Wkvb
    k_gemm_bf16<<<dim3((Hh * KVD + 127) / 128, BS / 128), 256, 0, stream>>>(
        kv_bf, wkvb_t, kv_f, BS, Hh * KVD, RKV);

    // 6) RoPE + layout: q_final, k_final, V^T (all bf16)
    k_assemble_q <<<BS * Hh, QKD, 0, stream>>>(q_f, qfin);
    k_assemble_kv<<<BS * Hh, 256, 0, stream>>>(kv_f, kva_f, kfin, vt);

    // 7) causal flash attention -> attn bf16 [BS, H*DV]
    k_mla_attention<<<Bb * Hh * 8, 256, 0, stream>>>(qfin, kfin, vt, attnb);

    // 8) out = attn @ Wout (fp32 output)
    k_gemm_bf16<<<dim3((Dd + 127) / 128, BS / 128), 256, 0, stream>>>(
        attnb, wout_t, out, BS, Dd, Hh * DV);
}
